// GCNNet_64295660421274
// MI455X (gfx1250) — compile-verified, matrix operand order checked by reading, not run on hardware
//
#include <hip/hip_runtime.h>
#include <hip/hip_bf16.h>

typedef __attribute__((ext_vector_type(16))) __bf16 v16bf;
typedef __attribute__((ext_vector_type(8)))  __bf16 v8bf;
typedef __attribute__((ext_vector_type(8)))  float  v8f;

#define CH 128   // hidden channels

// ---------------- utility kernels ----------------

__global__ void zero_f32(float* p, int n) {
    int i = blockIdx.x * blockDim.x + threadIdx.x;
    if (i < n) p[i] = 0.0f;
}

__global__ void deg_init(float* deg, int n) {
    int i = blockIdx.x * blockDim.x + threadIdx.x;
    if (i < n) deg[i] = 1.0f;            // self-loop
}

__global__ void deg_acc(const int* __restrict__ dst, float* deg, int e) {
    int i = blockIdx.x * blockDim.x + threadIdx.x;
    if (i < e) atomicAdd(&deg[dst[i]], 1.0f);
}

__global__ void deg_rsqrt(float* deg, int n) {
    int i = blockIdx.x * blockDim.x + threadIdx.x;
    if (i < n) deg[i] = rsqrtf(deg[i]);  // deg -> dis in place
}

__global__ void cvt_f32_bf16(const float* __restrict__ in, __bf16* __restrict__ out, int n) {
    int i = blockIdx.x * blockDim.x + threadIdx.x;
    if (i < n) out[i] = (__bf16)in[i];
}

// Pack weight matrix W[K=128][ncols] (row-major fp32) into per-lane WMMA B fragments (bf16).
// Fragment element e of lane L in tile (kt,nt):
//   K = kt*32 + (e&7) + ((e>>3)<<4) + ((L>>4)<<3),  Ncol = nt*16 + (L&15)
__global__ void pack_w(const float* __restrict__ W, __bf16* __restrict__ Bp, int ncols) {
    int ntiles = ncols >> 4;
    int total  = 4 * ntiles * 32 * 16;
    int idx = blockIdx.x * blockDim.x + threadIdx.x;
    if (idx >= total) return;
    int e    = idx & 15;
    int lane = (idx >> 4) & 31;
    int t    = idx >> 9;             // kt*ntiles + nt
    int nt   = t % ntiles;
    int kt   = t / ntiles;
    int k = kt * 32 + (e & 7) + ((e >> 3) << 4) + ((lane >> 4) << 3);
    int n = nt * 16 + (lane & 15);
    Bp[idx] = (__bf16)W[k * ncols + n];
}

// ---------------- WMMA GEMM: C[M x ncols] = A[M x 128](bf16) * Bpacked ----------------
// One block = ntiles waves; each wave owns one 16-col tile; block owns a 16-row tile.
// Epilogue: optional per-row scale (GCN dis), optional per-col bias + relu (MLP).
__global__ void gemm_wmma_bf16(const __bf16* __restrict__ A,
                               const __bf16* __restrict__ Bp,
                               float* __restrict__ C,
                               int M, int ntiles, int ldc,
                               const float* __restrict__ rowscale,
                               const float* __restrict__ bias, int relu) {
    int lane = threadIdx.x & 31;
    int nt   = threadIdx.x >> 5;
    int m0   = blockIdx.x * 16;
    int half = lane >> 4;
    int r    = lane & 15;

    int arow_idx = m0 + r;
    if (arow_idx >= M) arow_idx = M - 1;                 // clamp (loads only)
    const __bf16* arow = A + (size_t)arow_idx * CH;

    v8f acc = {};
#pragma unroll
    for (int kt = 0; kt < 4; ++kt) {
        int kb = kt * 32 + 8 * half;
        v8bf a0 = *(const v8bf*)(arow + kb);             // K = kb..kb+7
        v8bf a1 = *(const v8bf*)(arow + kb + 16);        // K = kb+16..kb+23
        v16bf a = __builtin_shufflevector(a0, a1, 0, 1, 2, 3, 4, 5, 6, 7,
                                                  8, 9, 10, 11, 12, 13, 14, 15);
        v16bf b = *(const v16bf*)(Bp + ((size_t)(kt * ntiles + nt) * 32 + lane) * 16);
        acc = __builtin_amdgcn_wmma_f32_16x16x32_bf16(false, a, false, b,
                                                      (short)0, acc, false, false);
    }

    int col = nt * 16 + (lane & 15);
    float bv = bias ? bias[col] : 0.0f;
#pragma unroll
    for (int v = 0; v < 8; ++v) {
        int m = m0 + v + 8 * half;
        if (m >= M) continue;
        float val = acc[v];
        if (rowscale) val *= rowscale[m];
        val += bv;
        if (relu) val = fmaxf(val, 0.0f);
        C[(size_t)m * ldc + col] = val;
    }
}

// ---------------- edge scatter: agg[dst] += hs[src], one wave per edge ----------------
__global__ void scatter_edges(const int* __restrict__ src, const int* __restrict__ dst,
                              const float* __restrict__ hs, float* __restrict__ agg, int e) {
    int idx  = blockIdx.x * blockDim.x + threadIdx.x;
    int ed   = idx >> 5;
    int lane = idx & 31;
    if (ed >= e) return;
    int s = src[ed], d = dst[ed];
    float4 v = *(const float4*)(hs + (size_t)s * CH + lane * 4);
    float* o = agg + (size_t)d * CH + lane * 4;
    atomicAdd(o + 0, v.x);
    atomicAdd(o + 1, v.y);
    atomicAdd(o + 2, v.z);
    atomicAdd(o + 3, v.w);
}

// ---------------- finalize: out = dis*(agg + hs) + b; accumulate BN stats ----------------
// block = 256 threads = 2 nodes x 128 channels per iteration
__global__ void finalize_stats(float* __restrict__ agg, const float* __restrict__ hs,
                               const float* __restrict__ dis, const float* __restrict__ bias,
                               float* __restrict__ sums,   // [0..127]=sum, [128..255]=sumsq
                               int n, int nodes_per_block) {
    int c   = threadIdx.x & 127;
    int sub = threadIdx.x >> 7;   // 0 or 1
    int base = blockIdx.x * nodes_per_block;
    int end  = base + nodes_per_block;
    if (end > n) end = n;
    float s = 0.0f, s2 = 0.0f;
    float bv = bias[c];
    for (int i = base + sub; i < end; i += 2) {
        size_t off = (size_t)i * CH + c;
        float v = dis[i] * (agg[off] + hs[off]) + bv;
        agg[off] = v;
        s  += v;
        s2 += v * v;
    }
    __shared__ float ls[256], ls2[256];
    ls[threadIdx.x]  = s;
    ls2[threadIdx.x] = s2;
    __syncthreads();
    if (sub == 0) {
        s  += ls[threadIdx.x + 128];
        s2 += ls2[threadIdx.x + 128];
        atomicAdd(&sums[c], s);
        atomicAdd(&sums[128 + c], s2);
    }
}

// ---------------- BN (training stats) + ReLU + convert to bf16 ----------------
__global__ void bn_relu_cvt(const float* __restrict__ in, const float* __restrict__ sums,
                            const float* __restrict__ gamma, const float* __restrict__ beta,
                            __bf16* __restrict__ out, int total, float inv_n) {
    int idx = blockIdx.x * blockDim.x + threadIdx.x;
    if (idx >= total) return;
    int c = idx & 127;
    float m   = sums[c] * inv_n;
    float var = sums[128 + c] * inv_n - m * m;
    float inv = rsqrtf(var + 1e-5f);
    float v = gamma[c] * (in[idx] - m) * inv + beta[c];
    out[idx] = (__bf16)fmaxf(v, 0.0f);
}

// ---------------- final 64-dot head: out[i] = m1[i,:] . Wm2 + bm2 ----------------
__global__ void mlp2(const float* __restrict__ m1, const float* __restrict__ Wm2,
                     const float* __restrict__ bm2, float* __restrict__ out, int n) {
    int i = blockIdx.x * blockDim.x + threadIdx.x;
    if (i >= n) return;
    float acc = bm2[0];
    const float4* p = (const float4*)(m1 + (size_t)i * 64);
    const float4* w = (const float4*)Wm2;
#pragma unroll
    for (int j = 0; j < 16; ++j) {
        float4 a = p[j], b = w[j];
        acc += a.x * b.x + a.y * b.y + a.z * b.z + a.w * b.w;
    }
    out[i] = acc;
}

// ---------------- launch ----------------

extern "C" void kernel_launch(void* const* d_in, const int* in_sizes, int n_in,
                              void* d_out, int out_size, void* d_ws, size_t ws_size,
                              hipStream_t stream) {
    const float* x   = (const float*)d_in[0];
    const int*   ei  = (const int*)d_in[1];
    const float* W1  = (const float*)d_in[2];
    const float* b1  = (const float*)d_in[3];
    const float* g1  = (const float*)d_in[4];
    const float* be1 = (const float*)d_in[5];
    const float* W2  = (const float*)d_in[6];
    const float* b2  = (const float*)d_in[7];
    const float* g2  = (const float*)d_in[8];
    const float* be2 = (const float*)d_in[9];
    const float* Wm1 = (const float*)d_in[10];
    const float* bm1 = (const float*)d_in[11];
    const float* Wm2 = (const float*)d_in[12];
    const float* bm2 = (const float*)d_in[13];
    float* out = (float*)d_out;

    const int n = in_sizes[0] / CH;       // 100000
    const int e = in_sizes[1] / 2;        // 1600000
    const int* src = ei;
    const int* dst = ei + e;

    // workspace carve (256B aligned)
    char* p = (char*)d_ws;
    size_t off = 0;
    auto carve = [&](size_t bytes) -> char* {
        char* q = p + off;
        off = (off + bytes + 255) & ~(size_t)255;
        return q;
    };
    __bf16* B0  = (__bf16*)carve((size_t)n * CH * sizeof(__bf16)); // bf16 activations
    float*  B1  = (float*) carve((size_t)n * CH * sizeof(float));  // hs = (A@W)*dis
    float*  B2  = (float*) carve((size_t)n * CH * sizeof(float));  // agg / layer output
    float*  dis = (float*) carve((size_t)n * sizeof(float));
    float*  sums = (float*)carve(256 * sizeof(float));
    __bf16* Bp1 = (__bf16*)carve(128 * 128 * sizeof(__bf16));
    __bf16* Bp2 = (__bf16*)carve(128 * 128 * sizeof(__bf16));
    __bf16* Bpm = (__bf16*)carve(128 * 64  * sizeof(__bf16));
    float*  M1  = (float*) carve((size_t)n * 64 * sizeof(float));
    (void)ws_size;

    const int TB = 256;
    const int nch = n * CH;
    const int mtiles = (n + 15) / 16;
    const float inv_n = 1.0f / (float)n;
    const int fin_blocks = (n + 255) / 256;

    // degrees -> dis
    deg_init<<<(n + TB - 1) / TB, TB, 0, stream>>>(dis, n);
    deg_acc<<<(e + TB - 1) / TB, TB, 0, stream>>>(dst, dis, e);
    deg_rsqrt<<<(n + TB - 1) / TB, TB, 0, stream>>>(dis, n);

    // pack weights, convert input
    pack_w<<<(16384 + TB - 1) / TB, TB, 0, stream>>>(W1, Bp1, 128);
    pack_w<<<(16384 + TB - 1) / TB, TB, 0, stream>>>(W2, Bp2, 128);
    pack_w<<<(8192 + TB - 1) / TB, TB, 0, stream>>>(Wm1, Bpm, 64);
    cvt_f32_bf16<<<(nch + TB - 1) / TB, TB, 0, stream>>>(x, B0, nch);

    // ---- layer 1 ----
    gemm_wmma_bf16<<<mtiles, 256, 0, stream>>>(B0, Bp1, B1, n, 8, CH, dis, nullptr, 0);
    zero_f32<<<(nch + TB - 1) / TB, TB, 0, stream>>>(B2, nch);
    zero_f32<<<1, 256, 0, stream>>>(sums, 256);
    {
        long long tot = (long long)e * 32;
        scatter_edges<<<(unsigned)((tot + TB - 1) / TB), TB, 0, stream>>>(src, dst, B1, B2, e);
    }
    finalize_stats<<<fin_blocks, 256, 0, stream>>>(B2, B1, dis, b1, sums, n, 256);
    bn_relu_cvt<<<(nch + TB - 1) / TB, TB, 0, stream>>>(B2, sums, g1, be1, B0, nch, inv_n);

    // ---- layer 2 ----
    gemm_wmma_bf16<<<mtiles, 256, 0, stream>>>(B0, Bp2, B1, n, 8, CH, dis, nullptr, 0);
    zero_f32<<<(nch + TB - 1) / TB, TB, 0, stream>>>(B2, nch);
    zero_f32<<<1, 256, 0, stream>>>(sums, 256);
    {
        long long tot = (long long)e * 32;
        scatter_edges<<<(unsigned)((tot + TB - 1) / TB), TB, 0, stream>>>(src, dst, B1, B2, e);
    }
    finalize_stats<<<fin_blocks, 256, 0, stream>>>(B2, B1, dis, b2, sums, n, 256);
    bn_relu_cvt<<<(nch + TB - 1) / TB, TB, 0, stream>>>(B2, sums, g2, be2, B0, nch, inv_n);

    // ---- MLP head ----
    gemm_wmma_bf16<<<mtiles, 128, 0, stream>>>(B0, Bpm, M1, n, 4, 64, nullptr, bm1, 1);
    mlp2<<<(n + TB - 1) / TB, TB, 0, stream>>>(M1, Wm2, bm2, out, n);
}
